// CNN_moe_noise_3504693313942
// MI455X (gfx1250) — matrix operand order repaired and checked by reading, not compiled, not run on hardware
//
#include <hip/hip_runtime.h>
#include <hip/hip_bf16.h>
#include <cstdint>

// ---------------------------------------------------------------------------
// Problem constants (from the reference)
// ---------------------------------------------------------------------------
#define BB   2048   // batch
#define EE   8      // experts
#define CC   100    // classes
#define HH   1024   // expert hidden
#define GG   2048   // gate hidden
#define DD   2048   // feat dim

typedef __attribute__((ext_vector_type(16))) _Float16 v16h;
typedef __attribute__((ext_vector_type(8)))  float    v8f;
typedef __attribute__((ext_vector_type(4)))  int      v4i;

union AFragU { v16h h; unsigned int u[8]; };
union BFragU { v16h h; unsigned int u[8]; };

// gfx1250 async global->LDS path (ASYNCcnt), if the toolchain exposes it.
// Probe round confirmed: param0 = v4i in addrspace(1) (non-const),
// param1 = addrspace(3) pointer, two trailing immediates.
#if __has_builtin(__builtin_amdgcn_global_load_async_to_lds_b128) && \
    __has_builtin(__builtin_amdgcn_s_wait_asynccnt)
#define USE_ASYNC_LDS 1
typedef __attribute__((address_space(1))) v4i GlobV4;
typedef __attribute__((address_space(3))) v4i LdsV4;
#endif

// ===========================================================================
// Generic WMMA GEMM:  C[M,N] = act(A[M,K] * W[K,N] + bias)
//   A: f16 row-major; W: f32 row-major [K,N] -> f16 LDS (transposed stage).
//   f32 accumulation via v_wmma_f32_16x16x32_f16. grid.z batches experts.
// Tile: BM=128, BN=128, BK=32; 256 threads = 8 waves; wave -> 32x64 subtile.
// ===========================================================================
template <typename TIn, typename TOut, bool RELU>
__global__ __launch_bounds__(256)
void gemm_wmma(const TIn* __restrict__ A, const float* __restrict__ W,
               const float* __restrict__ bias, TOut* __restrict__ Cmat,
               int M, int N, int Kd,
               long long sA, long long sW, long long sBias, long long sC)
{
    const int z = blockIdx.z;
    A    += (long long)z * sA;
    W    += (long long)z * sW;
    Cmat += (long long)z * sC;
    const float* biasz = bias ? bias + (long long)z * sBias : nullptr;

    __shared__ _Float16 As[128][40];   // [m][k]
    __shared__ _Float16 Bs[128][40];   // [n][k]

    const int tid  = threadIdx.x;
    const int gm0  = blockIdx.y * 128;
    const int gn0  = blockIdx.x * 128;
    const int wave = tid >> 5;
    const int lane = tid & 31;
    const int wm   = (wave & 3) * 32;   // 4 waves along M
    const int wn   = (wave >> 2) * 64;  // 2 waves along N

    v8f acc[2][4];
#pragma unroll
    for (int i = 0; i < 2; ++i)
#pragma unroll
        for (int j = 0; j < 4; ++j)
#pragma unroll
            for (int r = 0; r < 8; ++r) acc[i][j][r] = 0.0f;

    const int arow  = tid >> 1;          // A stage: 128 rows x 32 k
    const int akoff = (tid & 1) * 16;
    const int bk    = tid >> 3;          // B stage: 32 k x 128 n (coalesced in n)
    const int bnoff = (tid & 7) * 16;

    for (int k0 = 0; k0 < Kd; k0 += 32) {
        // ---- stage A tile (f16, contiguous 32B per thread) ----------------
        {
            const int gm   = gm0 + arow;
            const TIn* src = A + (long long)gm * Kd + k0 + akoff;
#if defined(USE_ASYNC_LDS)
            if constexpr (sizeof(TIn) == 2) {
                if (gm < M) {
                    const uintptr_t g = (uintptr_t)src;
                    const uintptr_t l = (uintptr_t)(unsigned)(uintptr_t)&As[arow][akoff];
                    __builtin_amdgcn_global_load_async_to_lds_b128(
                        (GlobV4*)g, (LdsV4*)l, 0, 0);
                    __builtin_amdgcn_global_load_async_to_lds_b128(
                        (GlobV4*)(g + 16), (LdsV4*)(l + 16), 0, 0);
                } else {
#pragma unroll
                    for (int i = 0; i < 16; ++i)
                        As[arow][akoff + i] = (_Float16)0.0f;
                }
            } else
#endif
            {
                if (gm < M) {
#pragma unroll
                    for (int i = 0; i < 16; ++i)
                        As[arow][akoff + i] = (_Float16)(float)src[i];
                    if (k0 + 32 < Kd) __builtin_prefetch(src + 32, 0, 0);
                } else {
#pragma unroll
                    for (int i = 0; i < 16; ++i)
                        As[arow][akoff + i] = (_Float16)0.0f;
                }
            }
        }
        // ---- stage B tile transposed: coalesced n-runs at fixed k ---------
        {
            const int k = k0 + bk;
            const float* wrow = W + (long long)k * N;
#pragma unroll
            for (int i = 0; i < 16; ++i) {
                const int gn = gn0 + bnoff + i;
                Bs[bnoff + i][bk] = (gn < N) ? (_Float16)wrow[gn] : (_Float16)0.0f;
            }
        }
#if defined(USE_ASYNC_LDS)
        if constexpr (sizeof(TIn) == 2) __builtin_amdgcn_s_wait_asynccnt(0);
#endif
        __syncthreads();

        // ---- fragments per ISA VGPR layouts -------------------------------
        AFragU af[2];
        {
            const int mrow = lane & 15;
            const int kl   = (lane >> 4) * 8;
#pragma unroll
            for (int i = 0; i < 2; ++i) {
                const int r = wm + i * 16 + mrow;
#pragma unroll
                for (int v = 0; v < 4; ++v) {
                    af[i].u[v]     = *(const unsigned int*)&As[r][kl + 2 * v];
                    af[i].u[v + 4] = *(const unsigned int*)&As[r][kl + 16 + 2 * v];
                }
            }
        }
        BFragU bf[4];
        {
            const int nc = lane & 15;
            const int kb = (lane >> 4) * 8;
#pragma unroll
            for (int j = 0; j < 4; ++j) {
                const int n = wn + j * 16 + nc;
#pragma unroll
                for (int v = 0; v < 8; ++v)
                    bf[j].u[v] = *(const unsigned int*)&Bs[n][kb + 2 * v + (v >> 2) * 8];
            }
        }
#pragma unroll
        for (int i = 0; i < 2; ++i)
#pragma unroll
            for (int j = 0; j < 4; ++j)
                acc[i][j] = __builtin_amdgcn_wmma_f32_16x16x32_f16(
                    false, af[i].h, false, bf[j].h, (short)0, acc[i][j], false, false);
        __syncthreads();
    }

    // ---- epilogue: D layout M = r + 8*(lane/16), N = lane%16 --------------
    const int ec = lane & 15;
    const int er = (lane >> 4) * 8;
#pragma unroll
    for (int j = 0; j < 4; ++j) {
        const int gn = gn0 + wn + j * 16 + ec;
        if (gn >= N) continue;
        const float bv = biasz ? biasz[gn] : 0.0f;
#pragma unroll
        for (int i = 0; i < 2; ++i) {
#pragma unroll
            for (int r = 0; r < 8; ++r) {
                const int gm = gm0 + wm + i * 16 + er + r;
                if (gm < M) {
                    float v = acc[i][j][r] + bv;
                    if (RELU) v = fmaxf(v, 0.0f);
                    Cmat[(long long)gm * N + gn] = (TOut)v;
                }
            }
        }
    }
}

// ===========================================================================
// Implicit-GEMM conv 3x3 SAME + folded BN + ReLU (WMMA f16).
//   M = B*H*W pixels, N = Cout, K = Cin*9 (padded to mult of 32).
//   Spatial dims are powers of two -> shift/mask address math.
// ===========================================================================
template <typename TIn>
__global__ __launch_bounds__(256)
void conv_wmma(const TIn* __restrict__ in, const float* __restrict__ w,
               const float* __restrict__ cb, const float* __restrict__ gam,
               const float* __restrict__ bet, _Float16* __restrict__ out,
               int Bn, int Cin, int Cout, int hwSh, int wSh)
{
    const int H  = 1 << (hwSh - wSh);
    const int W_ = 1 << wSh;
    const int HW = 1 << hwSh;
    const int M     = Bn << hwSh;
    const int Kreal = Cin * 9;
    const int Kpad  = (Kreal + 31) & ~31;

    __shared__ _Float16 As[128][40];
    __shared__ _Float16 Bs[128][40];

    const int tid  = threadIdx.x;
    const int gm0  = blockIdx.y * 128;
    const int gn0  = blockIdx.x * 128;
    const int wave = tid >> 5;
    const int lane = tid & 31;
    const int wm   = (wave & 3) * 32;
    const int wn   = (wave >> 2) * 64;

    v8f acc[2][4];
#pragma unroll
    for (int i = 0; i < 2; ++i)
#pragma unroll
        for (int j = 0; j < 4; ++j)
#pragma unroll
            for (int r = 0; r < 8; ++r) acc[i][j][r] = 0.0f;

    // im2col coordinates (fixed across the K loop)
    const int arow  = tid >> 1;
    const int akoff = (tid & 1) * 16;
    const int agm   = gm0 + arow;
    const int ab    = agm >> hwSh;
    const int ap    = agm & (HW - 1);
    const int ay    = ap >> wSh;
    const int ax    = ap & (W_ - 1);

    const int brow  = tid >> 1;          // B stage: w rows contiguous in k
    const int bkoff = (tid & 1) * 16;

    for (int k0 = 0; k0 < Kpad; k0 += 32) {
        // ---- A stage: im2col gather + f16 convert -------------------------
#pragma unroll
        for (int i = 0; i < 16; ++i) {
            const int k = k0 + akoff + i;
            _Float16 val = (_Float16)0.0f;
            if (agm < M && k < Kreal) {
                const int ci = k / 9;
                const int r9 = k - ci * 9;
                const int ky = r9 / 3;
                const int kx = r9 - ky * 3;
                const int iy = ay + ky - 1;
                const int ix = ax + kx - 1;
                if (iy >= 0 && iy < H && ix >= 0 && ix < W_)
                    val = (_Float16)(float)
                        in[(((long long)(ab * Cin + ci)) << hwSh) + (iy << wSh) + ix];
            }
            As[arow][akoff + i] = val;
        }
        // ---- B stage: weight rows (OIHW: contiguous in k) -----------------
        {
            const int gn = gn0 + brow;
#pragma unroll
            for (int i = 0; i < 16; ++i) {
                const int k = k0 + bkoff + i;
                Bs[brow][bkoff + i] = (gn < Cout && k < Kreal)
                    ? (_Float16)w[(long long)gn * Kreal + k] : (_Float16)0.0f;
            }
        }
        __syncthreads();

        AFragU af[2];
        {
            const int mrow = lane & 15;
            const int kl   = (lane >> 4) * 8;
#pragma unroll
            for (int i = 0; i < 2; ++i) {
                const int r = wm + i * 16 + mrow;
#pragma unroll
                for (int v = 0; v < 4; ++v) {
                    af[i].u[v]     = *(const unsigned int*)&As[r][kl + 2 * v];
                    af[i].u[v + 4] = *(const unsigned int*)&As[r][kl + 16 + 2 * v];
                }
            }
        }
        BFragU bf[4];
        {
            const int nc = lane & 15;
            const int kb = (lane >> 4) * 8;
#pragma unroll
            for (int j = 0; j < 4; ++j) {
                const int n = wn + j * 16 + nc;
#pragma unroll
                for (int v = 0; v < 8; ++v)
                    bf[j].u[v] = *(const unsigned int*)&Bs[n][kb + 2 * v + (v >> 2) * 8];
            }
        }
#pragma unroll
        for (int i = 0; i < 2; ++i)
#pragma unroll
            for (int j = 0; j < 4; ++j)
                acc[i][j] = __builtin_amdgcn_wmma_f32_16x16x32_f16(
                    false, af[i].h, false, bf[j].h, (short)0, acc[i][j], false, false);
        __syncthreads();
    }

    // ---- epilogue: (acc + cb) * g/sqrt(1+eps) + be, relu, NCHW ------------
    const int ec = lane & 15;
    const int er = (lane >> 4) * 8;
#pragma unroll
    for (int j = 0; j < 4; ++j) {
        const int gn = gn0 + wn + j * 16 + ec;
        if (gn >= Cout) continue;
        const float scale = gam[gn] * rsqrtf(1.0f + 1e-5f);
        const float cbv   = cb[gn];
        const float bev   = bet[gn];
#pragma unroll
        for (int i = 0; i < 2; ++i) {
#pragma unroll
            for (int r = 0; r < 8; ++r) {
                const int gm = gm0 + wm + i * 16 + er + r;
                if (gm < M) {
                    const int b = gm >> hwSh;
                    const int p = gm & (HW - 1);
                    float v = (acc[i][j][r] + cbv) * scale + bev;
                    v = fmaxf(v, 0.0f);
                    out[(((long long)(b * Cout + gn)) << hwSh) + p] = (_Float16)v;
                }
            }
        }
    }
}

// ===========================================================================
// 2x2 max-pool stride 2, f16 NCHW (pure shift/mask addressing)
// ===========================================================================
__global__ void pool2x2(const _Float16* __restrict__ in, _Float16* __restrict__ out,
                        int total, int hoSh, int woSh)
{
    const int t = blockIdx.x * 256 + threadIdx.x;
    if (t >= total) return;
    const int xo = t & ((1 << woSh) - 1);
    const int yo = (t >> woSh) & ((1 << hoSh) - 1);
    const long long bc = (unsigned)t >> (woSh + hoSh);
    const int wiSh = woSh + 1;
    const long long base = (bc << (hoSh + 1 + wiSh)) + ((long long)(yo * 2) << wiSh) + xo * 2;
    const int Win = 1 << wiSh;
    const float m = fmaxf(fmaxf((float)in[base], (float)in[base + 1]),
                          fmaxf((float)in[base + Win], (float)in[base + Win + 1]));
    out[t] = (_Float16)m;
}

// ===========================================================================
// Noisy top-k gating
// ===========================================================================
__device__ __forceinline__ float ncdf(float z)
{
    return 0.5f * (1.0f + erff(z * 0.70710678118654752f));
}

__global__ void gating_kernel(const float* __restrict__ clean,
                              const float* __restrict__ rawstd,
                              const float* __restrict__ noise,
                              float* __restrict__ gates,
                              float* __restrict__ importance,
                              float* __restrict__ loadv)
{
    const int b = blockIdx.x * blockDim.x + threadIdx.x;
    if (b >= BB) return;
    float cl[EE], st[EE], nz[EE];
#pragma unroll
    for (int e = 0; e < EE; ++e) {
        cl[e] = clean[b * EE + e];
        const float rs = rawstd[b * EE + e];
        const float sp = (rs > 20.0f) ? rs : log1pf(expf(rs));   // softplus
        st[e] = sp + 0.01f;
        nz[e] = cl[e] + noise[b * EE + e] * st[e];
    }
    float v0 = -1e30f, v1 = -1e30f, v2 = -1e30f;
    int   i0 = 0, i1 = 0;
#pragma unroll
    for (int e = 0; e < EE; ++e) {
        const float v = nz[e];
        if (v > v0)      { v2 = v1; v1 = v0; i1 = i0; v0 = v; i0 = e; }
        else if (v > v1) { v2 = v1; v1 = v; i1 = e; }
        else if (v > v2) { v2 = v; }
    }
    const float e1 = expf(v1 - v0);
    const float s  = 1.0f + e1;
    const float g0 = 1.0f / s, g1 = e1 / s;
#pragma unroll
    for (int e = 0; e < EE; ++e) gates[b * EE + e] = 0.0f;
    gates[b * EE + i0] = g0;
    gates[b * EE + i1] = g1;
    atomicAdd(&importance[i0], g0);
    atomicAdd(&importance[i1], g1);
    const float th_in = v2, th_out = v1;
#pragma unroll
    for (int e = 0; e < EE; ++e) {
        const float p = (nz[e] > th_in) ? ncdf((cl[e] - th_in) / st[e])
                                        : ncdf((cl[e] - th_out) / st[e]);
        atomicAdd(&loadv[e], p);
    }
}

// ===========================================================================
// Combine: y[b,c] = sum_e gates[b,e] * eo[e,b,c]
// ===========================================================================
__global__ void combine_kernel(const float* __restrict__ gates,
                               const float* __restrict__ eo,
                               float* __restrict__ y)
{
    const long long t = (long long)blockIdx.x * blockDim.x + threadIdx.x;
    if (t >= (long long)BB * CC) return;
    const int b = (int)(t / CC);
    const int c = (int)(t - (long long)b * CC);
    float s = 0.0f;
#pragma unroll
    for (int e = 0; e < EE; ++e)
        s += gates[b * EE + e] * eo[((long long)e * BB + b) * CC + c];
    y[t] = s;
}

// ===========================================================================
// Loss: cv2(importance) + cv2(load), ddof=1 over E=8
// ===========================================================================
__global__ void loss_kernel(const float* __restrict__ imp,
                            const float* __restrict__ loadv,
                            float* __restrict__ out)
{
    if (threadIdx.x != 0 || blockIdx.x != 0) return;
    float loss = 0.0f;
    for (int which = 0; which < 2; ++which) {
        const float* v = which ? loadv : imp;
        float m = 0.0f;
        for (int e = 0; e < EE; ++e) m += v[e];
        m *= (1.0f / EE);
        float var = 0.0f;
        for (int e = 0; e < EE; ++e) { const float d = v[e] - m; var += d * d; }
        var *= (1.0f / (EE - 1));
        loss += var / (m * m + 1e-10f);
    }
    out[(long long)BB * CC] = loss;
}

// ===========================================================================
// Host launcher
// ===========================================================================
extern "C" void kernel_launch(void* const* d_in, const int* in_sizes, int n_in,
                              void* d_out, int out_size, void* d_ws, size_t ws_size,
                              hipStream_t stream)
{
    (void)in_sizes; (void)n_in; (void)out_size; (void)ws_size;
    const float* x     = (const float*)d_in[0];
    const float* noise = (const float*)d_in[1];
    const float* cw[4] = { (const float*)d_in[2],  (const float*)d_in[6],
                           (const float*)d_in[10], (const float*)d_in[14] };
    const float* cb[4] = { (const float*)d_in[3],  (const float*)d_in[7],
                           (const float*)d_in[11], (const float*)d_in[15] };
    const float* gg[4] = { (const float*)d_in[4],  (const float*)d_in[8],
                           (const float*)d_in[12], (const float*)d_in[16] };
    const float* be[4] = { (const float*)d_in[5],  (const float*)d_in[9],
                           (const float*)d_in[13], (const float*)d_in[17] };
    const float* wg1  = (const float*)d_in[18]; const float* wg1b = (const float*)d_in[19];
    const float* wg2  = (const float*)d_in[20]; const float* wg2b = (const float*)d_in[21];
    const float* wn1  = (const float*)d_in[22]; const float* wn1b = (const float*)d_in[23];
    const float* wn2  = (const float*)d_in[24]; const float* wn2b = (const float*)d_in[25];
    const float* eW1  = (const float*)d_in[26]; const float* eb1  = (const float*)d_in[27];
    const float* eW2  = (const float*)d_in[28]; const float* eb2  = (const float*)d_in[29];
    const float* eW3  = (const float*)d_in[30]; const float* eb3  = (const float*)d_in[31];

    // ---- workspace carve-out ---------------------------------------------
    char* wsp = (char*)d_ws;
    size_t off = 0;
    auto alloc = [&](size_t bytes) -> void* {
        off = (off + 255) & ~(size_t)255;
        void* p = wsp + off;
        off += bytes;
        return p;
    };
    _Float16* cbuf = (_Float16*)alloc((size_t)BB * 64 * 32 * 32 * 2);  // conv out (reused)
    _Float16* p1   = (_Float16*)alloc((size_t)BB * 64 * 16 * 16 * 2);
    _Float16* p2   = (_Float16*)alloc((size_t)BB * 128 * 8 * 8 * 2);
    _Float16* p3   = (_Float16*)alloc((size_t)BB * 256 * 4 * 4 * 2);
    _Float16* feat = (_Float16*)alloc((size_t)BB * DD * 2);            // = pooled L4
    _Float16* gh   = (_Float16*)alloc((size_t)BB * GG * 2);
    _Float16* nh   = (_Float16*)alloc((size_t)BB * GG * 2);
    float* clean   = (float*)alloc((size_t)BB * EE * 4);
    float* rawstd  = (float*)alloc((size_t)BB * EE * 4);
    float* gates   = (float*)alloc((size_t)BB * EE * 4);
    float* imp     = (float*)alloc(2 * EE * 4);
    float* loadv   = imp + EE;
    _Float16* eh1  = (_Float16*)alloc((size_t)EE * BB * HH * 2);
    _Float16* eh2  = (_Float16*)alloc((size_t)EE * BB * (HH / 2) * 2);
    float* eo      = (float*)alloc((size_t)EE * BB * CC * 4);

    // ---- conv tower ------------------------------------------------------
    conv_wmma<float><<<dim3(1, (BB * 1024) / 128, 1), 256, 0, stream>>>(
        x, cw[0], cb[0], gg[0], be[0], cbuf, BB, 3, 64, /*hwSh*/10, /*wSh*/5);
    pool2x2<<<(BB * 64 * 256) / 256, 256, 0, stream>>>(cbuf, p1, BB * 64 * 256, 4, 4);

    conv_wmma<_Float16><<<dim3(1, (BB * 256) / 128, 1), 256, 0, stream>>>(
        p1, cw[1], cb[1], gg[1], be[1], cbuf, BB, 64, 128, 8, 4);
    pool2x2<<<(BB * 128 * 64) / 256, 256, 0, stream>>>(cbuf, p2, BB * 128 * 64, 3, 3);

    conv_wmma<_Float16><<<dim3(2, (BB * 64) / 128, 1), 256, 0, stream>>>(
        p2, cw[2], cb[2], gg[2], be[2], cbuf, BB, 128, 256, 6, 3);
    pool2x2<<<(BB * 256 * 16) / 256, 256, 0, stream>>>(cbuf, p3, BB * 256 * 16, 2, 2);

    conv_wmma<_Float16><<<dim3(4, (BB * 16) / 128, 1), 256, 0, stream>>>(
        p3, cw[3], cb[3], gg[3], be[3], cbuf, BB, 256, 512, 4, 2);
    pool2x2<<<(BB * 512 * 4) / 256, 256, 0, stream>>>(cbuf, feat, BB * 512 * 4, 1, 1);

    // ---- gating MLPs -----------------------------------------------------
    gemm_wmma<_Float16, _Float16, true><<<dim3(GG / 128, BB / 128, 1), 256, 0, stream>>>(
        feat, wg1, wg1b, gh, BB, GG, DD, 0, 0, 0, 0);
    gemm_wmma<_Float16, float, false><<<dim3(1, BB / 128, 1), 256, 0, stream>>>(
        gh, wg2, wg2b, clean, BB, EE, GG, 0, 0, 0, 0);
    gemm_wmma<_Float16, _Float16, true><<<dim3(GG / 128, BB / 128, 1), 256, 0, stream>>>(
        feat, wn1, wn1b, nh, BB, GG, DD, 0, 0, 0, 0);
    gemm_wmma<_Float16, float, false><<<dim3(1, BB / 128, 1), 256, 0, stream>>>(
        nh, wn2, wn2b, rawstd, BB, EE, GG, 0, 0, 0, 0);

    // ---- gating scalar stage ---------------------------------------------
    (void)hipMemsetAsync(imp, 0, 2 * EE * sizeof(float), stream);
    gating_kernel<<<BB / 256, 256, 0, stream>>>(clean, rawstd, noise, gates, imp, loadv);

    // ---- experts (batched over grid.z = 8) -------------------------------
    gemm_wmma<_Float16, _Float16, true><<<dim3(HH / 128, BB / 128, EE), 256, 0, stream>>>(
        feat, eW1, eb1, eh1, BB, HH, DD,
        0, (long long)DD * HH, HH, (long long)BB * HH);
    gemm_wmma<_Float16, _Float16, true><<<dim3((HH / 2) / 128, BB / 128, EE), 256, 0, stream>>>(
        eh1, eW2, eb2, eh2, BB, HH / 2, HH,
        (long long)BB * HH, (long long)HH * (HH / 2), HH / 2, (long long)BB * (HH / 2));
    gemm_wmma<_Float16, float, false><<<dim3(1, BB / 128, EE), 256, 0, stream>>>(
        eh2, eW3, eb3, eo, BB, CC, HH / 2,
        (long long)BB * (HH / 2), (long long)(HH / 2) * CC, CC, (long long)BB * CC);

    // ---- combine + loss --------------------------------------------------
    combine_kernel<<<(BB * CC + 255) / 256, 256, 0, stream>>>(gates, eo, (float*)d_out);
    loss_kernel<<<1, 1, 0, stream>>>(imp, loadv, (float*)d_out);
}
